// gru_74594991997179
// MI455X (gfx1250) — matrix operand order repaired and checked by reading, not compiled, not run on hardware
//
#include <hip/hip_runtime.h>

// ---------------------------------------------------------------------------
// Bidirectional GRU (H=32, B=128, T=2500, I=76) for gfx1250 (MI455X).
// Phase 1 (parallel, bandwidth-bound): XG[dir][t][b][g] = X @ W_ih^T + b_ih
//   via f16 WMMA (K padded 76->96, 3 chained v_wmma_f32_16x16x32_f16 per
//   gate tile), W_ih/b_ih fragments pinned in VGPRs, double-buffered X tile.
// Phase 2 (serial, latency-bound): 16 independent WGs (2 dirs x 8 batch
//   tiles). Per step: hg = h @ W_hh^T + b_hh as ONE WMMA per 16-gate tile
//   (bias folded into C operand), gates via v_exp_f32 + v_rcp_f32 only,
//   h kept in registers (f32) + LDS (f16 for the A fragment), xg prefetched
//   and double-buffered in LDS. 2 barriers per step.
// ---------------------------------------------------------------------------

typedef __attribute__((ext_vector_type(16))) _Float16 v16h;
typedef __attribute__((ext_vector_type(8)))  _Float16 v8h;
typedef __attribute__((ext_vector_type(8)))  float    v8f;
typedef __attribute__((ext_vector_type(4)))  float    v4f;

#define T_LEN 2500
#define BATCH 128
#define INP   76
#define HID   32
#define G3    96
#define TC    10

// sigma(x) = 1/(1+e^-x): one v_exp_f32 + one v_rcp_f32.
__device__ __forceinline__ float sigm_f(float x) {
  return __builtin_amdgcn_rcpf(1.0f + __expf(-x));
}
// tanh(x) = 2*sigma(2x) - 1 (saturates correctly through exp overflow).
__device__ __forceinline__ float tanh_f(float x) {
  return 2.0f * __builtin_amdgcn_rcpf(1.0f + __expf(-2.0f * x)) - 1.0f;
}

// ------------------------- Phase 1: input projection ------------------------
__global__ __launch_bounds__(256) void gru_xproj_wmma(
    const float* __restrict__ X,       // [B][T][76]
    const float* __restrict__ W_ih_f, const float* __restrict__ b_ih_f,
    const float* __restrict__ W_ih_b, const float* __restrict__ b_ih_b,
    float* __restrict__ XG)            // [2][T][B][96]
{
  __shared__ __align__(16) _Float16 Xh[2][16 * G3];   // X tile, f16, K padded

  const int tid  = threadIdx.x;
  const int nCh  = T_LEN / TC;                 // 250
  int bid = blockIdx.x;                        // dir*(nCh*8) + tc*8 + mb
  const int dir  = bid / (nCh * 8);
  int rem = bid % (nCh * 8);
  const int tc   = rem / 8;
  const int mb   = rem % 8;
  const int b0   = mb * 16;

  const float* W    = dir ? W_ih_b : W_ih_f;
  const float* bias = dir ? b_ih_b : b_ih_f;
  float* xgd = XG + (size_t)dir * T_LEN * BATCH * G3;

  const int lane  = tid & 31;
  const int wave  = tid >> 5;
  const int lhalf = (lane < 16) ? 0 : 1;
  const int lm    = lane & 15;

  // --- pinned B fragments (W_ih^T, 3 k-chunks of 32) + bias C fragment
  v16h bf[3] = {};
  v8f  cf = {};
  if (wave < 6) {
    const int g = wave * 16 + lm;
    const float bg = bias[g];
#pragma unroll
    for (int r = 0; r < 8; ++r) cf[r] = bg;
#pragma unroll
    for (int kc = 0; kc < 3; ++kc)
#pragma unroll
      for (int e = 0; e < 16; ++e) {
        int k = kc * 32 + lhalf * 8 + e + (e >= 8 ? 8 : 0);  // B 32x16: lanes<16 K{0-7,16-23}, else K{8-15,24-31}
        bf[kc][e] = (k < INP) ? (_Float16)W[g * INP + k] : (_Float16)0.0f;
      }
  }

  // --- zero the K-pad region of both buffers (written once, never touched)
  for (int j = tid; j < 16 * G3; j += 256) {
    if ((j % G3) >= INP) { Xh[0][j] = (_Float16)0.0f; Xh[1][j] = (_Float16)0.0f; }
  }
  // --- preload first X tile (valid region only)
  const int t0 = tc * TC;
  for (int j = tid; j < 16 * INP; j += 256) {
    int m = j / INP, i = j % INP;
    Xh[0][m * G3 + i] = (_Float16)X[((size_t)(b0 + m) * T_LEN + t0) * INP + i];
  }
  __syncthreads();

  for (int tt = 0; tt < TC; ++tt) {
    const int t   = t0 + tt;
    const int cur = tt & 1;

    // prefetch next X tile into registers (1216 = 4*256 + 192)
    float px[5];
    const bool have = (tt + 1 < TC);
    if (have) {
#pragma unroll
      for (int s = 0; s < 5; ++s) {
        int j = tid + s * 256;
        if (j < 16 * INP) {
          int m = j / INP, i = j % INP;
          px[s] = X[((size_t)(b0 + m) * T_LEN + (t + 1)) * INP + i];
        }
      }
    }

    if (wave < 6) {
      const v8h* xp = (const v8h*)Xh[cur];     // 12 v8h per 96-f16 row
      v8f acc = cf;
#pragma unroll
      for (int kc = 0; kc < 3; ++kc) {
        // A 16x32: lane<16 holds K{0-7,16-23}, lane>=16 K{8-15,24-31} (+kc*32)
        v8h alo = xp[lm * 12 + kc * 4 + lhalf];
        v8h ahi = xp[lm * 12 + kc * 4 + 2 + lhalf];
        v16h a;
#pragma unroll
        for (int e = 0; e < 8; ++e) { a[e] = alo[e]; a[e + 8] = ahi[e]; }
        acc = __builtin_amdgcn_wmma_f32_16x16x32_f16(
            false, a, false, bf[kc], (short)0, acc, false, false);
      }
      // D: element r -> (m = r + lhalf*8, g = wave*16 + lm)
      float* dst = xgd + ((size_t)t * BATCH + b0) * G3 + wave * 16 + lm;
#pragma unroll
      for (int r = 0; r < 8; ++r) dst[(size_t)(r + lhalf * 8) * G3] = acc[r];
    }

    if (have) {
#pragma unroll
      for (int s = 0; s < 5; ++s) {
        int j = tid + s * 256;
        if (j < 16 * INP) {
          int m = j / INP, i = j % INP;
          Xh[cur ^ 1][m * G3 + i] = (_Float16)px[s];
        }
      }
    }
    __syncthreads();    // one barrier per t: nxt-buffer writes vs next reads
  }
}

// ------------------------- Phase 2: recurrence ------------------------------
__global__ __launch_bounds__(256) void gru_recur_kernel(
    const float* __restrict__ XG,      // [2][T][B][96]
    const float* __restrict__ W_hh_f, const float* __restrict__ b_hh_f,
    const float* __restrict__ W_hh_b, const float* __restrict__ b_hh_b,
    float* __restrict__ out)           // [B][2][32]
{
  __shared__ __align__(16) _Float16 h16s[16 * HID];     // h (f16) for WMMA A
  __shared__ __align__(16) float    hgTs[G3 * 16];      // hg transposed [g][m]
  __shared__ __align__(16) float    xgbuf[2][16 * G3];  // double-buffered xg_t

  const int tid = threadIdx.x;
  const int wg  = blockIdx.x;           // 0..15
  const int dir = wg >> 3;
  const int b0  = (wg & 7) * 16;

  const float* xg_base = XG + (size_t)dir * T_LEN * BATCH * G3;
  const float* W_hh = dir ? W_hh_b : W_hh_f;
  const float* b_hh = dir ? b_hh_b : b_hh_f;

  const int lane  = tid & 31;
  const int wave  = tid >> 5;
  const int lhalf = (lane < 16) ? 0 : 1;
  const int lm    = lane & 15;

  // --- pinned W_hh B fragment + b_hh folded into the C fragment
  v16h bfrag = {};
  v8f  cfrag = {};
  if (wave < 6) {
    const int g = wave * 16 + lm;
    const float bg = b_hh[g];
#pragma unroll
    for (int r = 0; r < 8; ++r) cfrag[r] = bg;
#pragma unroll
    for (int e = 0; e < 16; ++e)
      bfrag[e] = (_Float16)W_hh[g * HID + lhalf * 8 + e + (e >= 8 ? 8 : 0)];
  }

  for (int j = tid; j < 16 * HID; j += 256) h16s[j] = (_Float16)0.0f;

  // each thread owns h for (m0,k0) and (m0,k0+16); f32 master stays in regs
  const int m0 = tid & 15;
  const int k0 = tid >> 4;              // 0..15
  float h0 = 0.0f, h1 = 0.0f;

  // --- preload xg for first step (1536 floats, coalesced)
  {
    const int tS = dir ? (T_LEN - 1) : 0;
    const float* src = xg_base + (size_t)tS * BATCH * G3 + (size_t)b0 * G3;
#pragma unroll
    for (int s = 0; s < 6; ++s) xgbuf[0][tid + s * 256] = src[tid + s * 256];
  }
  __syncthreads();

  int cur = 0;
  for (int step = 0; step < T_LEN; ++step) {
    const int t = dir ? (T_LEN - 1 - step) : step;

    // 1) global prefetch of next xg (off the serial chain)
    float pf[6];
    const bool havePf = (step + 1 < T_LEN);
    if (havePf) {
      const int tn = dir ? (t - 1) : (t + 1);
      const float* src = xg_base + (size_t)tn * BATCH * G3 + (size_t)b0 * G3;
#pragma unroll
      for (int s = 0; s < 6; ++s) pf[s] = src[tid + s * 256];
    }

    // 2) hg = h @ W_hh^T + b_hh : one WMMA per 16-gate tile
    if (wave < 6) {
      const v8h* hp = (const v8h*)h16s;
      v8h alo = hp[lm * 4 + lhalf];
      v8h ahi = hp[lm * 4 + 2 + lhalf];
      v16h a;
#pragma unroll
      for (int e = 0; e < 8; ++e) { a[e] = alo[e]; a[e + 8] = ahi[e]; }

      v8f d = __builtin_amdgcn_wmma_f32_16x16x32_f16(
          false, a, false, bfrag, (short)0, cfrag, false, false);

      float* dst = &hgTs[(wave * 16 + lm) * 16 + lhalf * 8];
      v4f dlo, dhi;
#pragma unroll
      for (int r = 0; r < 4; ++r) { dlo[r] = d[r]; dhi[r] = d[r + 4]; }
      *(v4f*)dst       = dlo;
      *(v4f*)(dst + 4) = dhi;
    }
    __syncthreads();

    // 3) gate math + h update (2 elements per thread, all in registers)
    {
      const float* xg = &xgbuf[cur][m0 * G3];
      float r = sigm_f(xg[k0]      + hgTs[(k0)      * 16 + m0]);
      float z = sigm_f(xg[32 + k0] + hgTs[(32 + k0) * 16 + m0]);
      float n = tanh_f(xg[64 + k0] + r * hgTs[(64 + k0) * 16 + m0]);
      h0 = (1.0f - z) * n + z * h0;
      h16s[m0 * HID + k0] = (_Float16)h0;

      const int k1 = k0 + 16;
      float r1 = sigm_f(xg[k1]      + hgTs[(k1)      * 16 + m0]);
      float z1 = sigm_f(xg[32 + k1] + hgTs[(32 + k1) * 16 + m0]);
      float n1 = tanh_f(xg[64 + k1] + r1 * hgTs[(64 + k1) * 16 + m0]);
      h1 = (1.0f - z1) * n1 + z1 * h1;
      h16s[m0 * HID + k1] = (_Float16)h1;
    }

    // 4) land prefetched xg into the other buffer
    if (havePf) {
#pragma unroll
      for (int s = 0; s < 6; ++s) xgbuf[cur ^ 1][tid + s * 256] = pf[s];
    }
    __syncthreads();
    cur ^= 1;
  }

  // --- final hidden state: out[b][dir][k]
  out[((size_t)(b0 + m0) * 2 + dir) * HID + k0]      = h0;
  out[((size_t)(b0 + m0) * 2 + dir) * HID + k0 + 16] = h1;
}

// ------------------------------- launcher -----------------------------------
extern "C" void kernel_launch(void* const* d_in, const int* in_sizes, int n_in,
                              void* d_out, int out_size, void* d_ws, size_t ws_size,
                              hipStream_t stream) {
  const float* X      = (const float*)d_in[0];
  const float* W_ih_f = (const float*)d_in[1];
  const float* W_hh_f = (const float*)d_in[2];
  const float* b_ih_f = (const float*)d_in[3];
  const float* b_hh_f = (const float*)d_in[4];
  const float* W_ih_b = (const float*)d_in[5];
  const float* W_hh_b = (const float*)d_in[6];
  const float* b_ih_b = (const float*)d_in[7];
  const float* b_hh_b = (const float*)d_in[8];

  float* XG = (float*)d_ws;   // needs 2*T*B*96*4 = ~246 MB of workspace

  const int nBlocksProj = 2 * (T_LEN / TC) * 8;   // 4000
  gru_xproj_wmma<<<nBlocksProj, 256, 0, stream>>>(
      X, W_ih_f, b_ih_f, W_ih_b, b_ih_b, XG);

  gru_recur_kernel<<<16, 256, 0, stream>>>(
      XG, W_hh_f, b_hh_f, W_hh_b, b_hh_b, (float*)d_out);
}
